// NTRDModel_10093173145844
// MI455X (gfx1250) — compile-verified, matrix operand order checked by reading, not run on hardware
//
#include <hip/hip_runtime.h>
#include <hip/hip_bf16.h>
#include <math.h>

// Problem constants (from reference)
#define B_SZ 2048
#define LE 100
#define LW 256
#define N_ENTITY 50000
#define N_WORD 30000
#define D_DIM 256

typedef _Float16 v16h __attribute__((ext_vector_type(16)));
typedef _Float16 v8h  __attribute__((ext_vector_type(8)));
typedef _Float16 v4h  __attribute__((ext_vector_type(4)));
typedef float    v8f  __attribute__((ext_vector_type(8)));
typedef float    v4f  __attribute__((ext_vector_type(4)));

typedef unsigned int u32x4 __attribute__((ext_vector_type(4)));
typedef int          i32x8 __attribute__((ext_vector_type(8)));
typedef int          i32x4 __attribute__((ext_vector_type(4)));

// ---------------------------------------------------------------------------
// Kernel 1: gather + mean pool. One block per batch row, one thread per dim.
// ---------------------------------------------------------------------------
__global__ __launch_bounds__(256) void pool_kernel(
    const float* __restrict__ entity_emb, const float* __restrict__ word_emb,
    const int* __restrict__ related_entity, const int* __restrict__ related_word,
    float* __restrict__ ent_rep, float* __restrict__ word_rep)
{
    const int b = blockIdx.x;
    const int d = threadIdx.x;

    const int* re = related_entity + (size_t)b * LE;
    float accE = 0.0f;
    for (int i = 0; i < LE; ++i) {
        int idx = re[i];
        accE += entity_emb[(size_t)idx * D_DIM + d];
    }
    ent_rep[(size_t)b * D_DIM + d] = accE * (1.0f / LE);

    const int* rw = related_word + (size_t)b * LW;
    float accW = 0.0f;
    for (int i = 0; i < LW; ++i) {
        int idx = rw[i];
        accW += word_emb[(size_t)idx * D_DIM + d];
    }
    word_rep[(size_t)b * D_DIM + d] = accW * (1.0f / LW);
}

// ---------------------------------------------------------------------------
// Kernel 2: GateLayer + user vector, fused f16 conversion of the GEMM A matrix.
// ---------------------------------------------------------------------------
__global__ __launch_bounds__(256) void gate_user_kernel(
    const float* __restrict__ ent_rep, const float* __restrict__ word_rep,
    const float* __restrict__ w1, const float* __restrict__ b1,
    const float* __restrict__ w2, const float* __restrict__ b2,
    _Float16* __restrict__ A_h)
{
    __shared__ float cat[2 * D_DIM];
    __shared__ float red[D_DIM];
    const int b = blockIdx.x;
    const int t = threadIdx.x;

    cat[t]         = ent_rep[(size_t)b * D_DIM + t];
    cat[D_DIM + t] = word_rep[(size_t)b * D_DIM + t];
    __syncthreads();

    float acc = b1[t];
    const float* wrow = w1 + (size_t)t * (2 * D_DIM);
    #pragma unroll 8
    for (int k = 0; k < 2 * D_DIM; ++k)
        acc = fmaf(cat[k], wrow[k], acc);

    red[t] = acc * w2[t];
    __syncthreads();
    for (int s = 128; s > 0; s >>= 1) {
        if (t < s) red[t] += red[t + s];
        __syncthreads();
    }
    const float gate = 1.0f / (1.0f + expf(-(red[0] + b2[0])));

    const float u = gate * cat[t] + (1.0f - gate) * cat[D_DIM + t];
    A_h[(size_t)b * D_DIM + t] = (_Float16)u;
}

// ---------------------------------------------------------------------------
// Kernel 3: f32 -> f16 conversion of entity_emb (B matrix of the big GEMM).
// ---------------------------------------------------------------------------
__global__ __launch_bounds__(256) void cvt_f16_kernel(
    const float* __restrict__ src, _Float16* __restrict__ dst, int n4)
{
    int i = blockIdx.x * blockDim.x + threadIdx.x;
    if (i < n4) {
        v4f x = *(const v4f*)(src + 4 * (size_t)i);
        *(v4h*)(dst + 4 * (size_t)i) = __builtin_convertvector(x, v4h);
    }
}

// ---------------------------------------------------------------------------
// Kernel 4: rec_scores = user @ entity_emb.T + rec_bias via WMMA f16->f32.
//
// Block = 8 waves = one 128(M) x 80(N) super-tile. The 80x256 f16 B strip
// (40 KB) is DMA'd into LDS ONCE per block by the Tensor Data Mover
// (tensor_load_to_lds, pad_enable inserts 16 B after every 512 B row ->
// LDS row stride 528 B, conflict-free ds_load_b128 banks), then shared by
// all 8 waves. Each wave owns one 16-row M-tile and 5 accumulator tiles.
// Inner loop: all 10 ds_load_b128 of a k-step are issued BEFORE the 5 WMMAs
// (separate load/compute phases) so DS latency overlaps matrix issue via
// partial s_wait_dscnt instead of serial wait-0 per WMMA.
// ---------------------------------------------------------------------------
#define N_GROUPS 625           // 50000 / 80
#define NT 5                   // 16-wide N tiles per wave
#define M_BLOCKS (B_SZ / 128)  // 16
#define LDS_ROW 528            // 512 B row + 16 B TDM pad
#define LDS_BYTES (80 * LDS_ROW)

__global__ __launch_bounds__(256) void wmma_gemm_kernel(
    const _Float16* __restrict__ A_h,   // [2048, 256]
    const _Float16* __restrict__ E_h,   // [50000, 256]
    const float*    __restrict__ rec_bias,
    float*          __restrict__ out)   // [2048, 50000]
{
    extern __shared__ char smem[];      // LDS_BYTES, starts at LDS offset 0

    const int w    = threadIdx.x >> 5;          // wave in block: 0..7
    const int lane = threadIdx.x & 31;
    const int bm   = blockIdx.x / N_GROUPS;     // M-block 0..15
    const int ng   = blockIdx.x - bm * N_GROUPS;// N-group 0..624
    const int n0   = ng * (NT * 16);

    const int lrow = lane & 15;
    const int ksel = lane >> 4;                 // 0 or 1

    // --- Stage B strip [80 x 256] f16 into LDS via TDM (wave 0 only) ------
    if (threadIdx.x < 32) {
        unsigned long long ga =
            (unsigned long long)(const void*)(E_h + (size_t)n0 * D_DIM);
        // D# group 0: count=1 | lds_addr=0 | global_addr | type=2
        u32x4 g0 = { 1u,
                     0u,
                     (unsigned)(ga & 0xFFFFFFFFu),
                     ((unsigned)((ga >> 32) & 0x01FFFFFFu)) | 0x80000000u };
        // D# group 1:
        // w0: data_size=1(2B) | pad_enable | pad_interval=6(128dw) | pad_amount=3(4dw)
        // w1: tensor_dim0=256 (<<16) ; w2: tensor_dim1=50000 (<<16)
        // w3: tile_dim0=256 (<<16)   ; w4: tile_dim1=80
        // w5: tensor_dim0_stride=256 ; w6,w7: dim1 stride unused
        i32x8 g1 = { (int)0x07910000u,
                     (int)(256u << 16),
                     (int)(50000u << 16),
                     (int)(256u << 16),
                     80, 256, 0, 0 };
        i32x4 z4 = { 0, 0, 0, 0 };
#if defined(__clang_major__) && (__clang_major__ >= 23)
        i32x8 z8 = { 0, 0, 0, 0, 0, 0, 0, 0 };
        __builtin_amdgcn_tensor_load_to_lds(g0, g1, z4, z4, z8, 0);
#else
        __builtin_amdgcn_tensor_load_to_lds(g0, g1, z4, z4, 0);
#endif
        __builtin_amdgcn_s_wait_tensorcnt(0);
    }
    __syncthreads();

    // --- Compute ----------------------------------------------------------
    const _Float16* Arow =
        A_h + ((size_t)(bm * 128 + w * 16 + lrow)) * D_DIM + ksel * 8;

    v8f acc[NT] = {};

    #pragma unroll 2
    for (int kk = 0; kk < 8; ++kk) {
        const int k0 = kk * 32;

        // Phase 1: issue A (global b128 x2) and all B fragments (ds b128 x10)
        v8h alo = *(const v8h*)(Arow + k0);
        v8h ahi = *(const v8h*)(Arow + k0 + 16);
        v16h a = __builtin_shufflevector(alo, ahi,
                 0,1,2,3,4,5,6,7,8,9,10,11,12,13,14,15);

        v16h bf[NT];
        #pragma unroll
        for (int j = 0; j < NT; ++j) {
            const char* bp = smem + (j * 16 + lrow) * LDS_ROW
                                  + k0 * 2 + ksel * 16;
            v8h blo = *(const v8h*)(bp);
            v8h bhi = *(const v8h*)(bp + 32);
            bf[j] = __builtin_shufflevector(blo, bhi,
                    0,1,2,3,4,5,6,7,8,9,10,11,12,13,14,15);
        }

        // Phase 2: matrix ops (waits can retire incrementally)
        #pragma unroll
        for (int j = 0; j < NT; ++j) {
            acc[j] = __builtin_amdgcn_wmma_f32_16x16x32_f16(
                         /*neg_a=*/false, a, /*neg_b=*/false, bf[j],
                         /*c_mod=*/(short)0, acc[j],
                         /*reuse_a=*/false, /*reuse_b=*/false);
        }
    }

    // --- Epilogue: VGPR r is row (r + 8*ksel), col = lrow within tile -----
    const size_t rowBase = (size_t)(bm * 128 + w * 16 + 8 * ksel) * N_ENTITY;
    #pragma unroll
    for (int j = 0; j < NT; ++j) {
        const int col = n0 + j * 16 + lrow;
        const float bias = rec_bias[col];
        #pragma unroll
        for (int r = 0; r < 8; ++r) {
            out[rowBase + (size_t)r * N_ENTITY + col] = acc[j][r] + bias;
        }
    }
}

// ---------------------------------------------------------------------------
// Kernel 5: per-row online logsumexp + label pick -> per-row loss partial.
// ---------------------------------------------------------------------------
__global__ __launch_bounds__(256) void loss_row_kernel(
    const float* __restrict__ scores, const int* __restrict__ labels,
    float* __restrict__ partial)
{
    __shared__ float sm[256];
    __shared__ float ss[256];
    const int b = blockIdx.x;
    const int t = threadIdx.x;
    const float* row = scores + (size_t)b * N_ENTITY;

    float m = -1e30f, s = 0.0f;
    for (int i = t; i < N_ENTITY; i += 256) {
        float x = row[i];
        float nm = fmaxf(m, x);
        s = s * expf(m - nm) + expf(x - nm);
        m = nm;
    }
    sm[t] = m; ss[t] = s;
    __syncthreads();
    for (int st = 128; st > 0; st >>= 1) {
        if (t < st) {
            float m2 = sm[t + st], s2 = ss[t + st];
            float nm = fmaxf(sm[t], m2);
            ss[t] = ss[t] * expf(sm[t] - nm) + s2 * expf(m2 - nm);
            sm[t] = nm;
        }
        __syncthreads();
    }
    if (t == 0) {
        float lse = sm[0] + logf(ss[0]);
        partial[b] = lse - row[labels[b]];
    }
}

// ---------------------------------------------------------------------------
// Kernel 6: deterministic reduction of the 2048 partials -> mean loss.
// ---------------------------------------------------------------------------
__global__ __launch_bounds__(256) void loss_reduce_kernel(
    const float* __restrict__ partial, float* __restrict__ out_loss)
{
    __shared__ float sh[256];
    const int t = threadIdx.x;
    float a = 0.0f;
    for (int i = t; i < B_SZ; i += 256) a += partial[i];
    sh[t] = a;
    __syncthreads();
    for (int s = 128; s > 0; s >>= 1) {
        if (t < s) sh[t] += sh[t + s];
        __syncthreads();
    }
    if (t == 0) *out_loss = sh[0] * (1.0f / B_SZ);
}

// ---------------------------------------------------------------------------
extern "C" void kernel_launch(void* const* d_in, const int* in_sizes, int n_in,
                              void* d_out, int out_size, void* d_ws, size_t ws_size,
                              hipStream_t stream) {
    const float* entity_emb     = (const float*)d_in[0];
    const float* word_emb       = (const float*)d_in[1];
    const float* w1             = (const float*)d_in[2];
    const float* b1             = (const float*)d_in[3];
    const float* w2             = (const float*)d_in[4];
    const float* b2             = (const float*)d_in[5];
    const float* rec_bias       = (const float*)d_in[6];
    const int*   related_entity = (const int*)d_in[7];
    const int*   related_word   = (const int*)d_in[8];
    const int*   labels         = (const int*)d_in[9];

    float* out_scores = (float*)d_out;                        // [2048*50000]
    float* out_loss   = out_scores + (size_t)B_SZ * N_ENTITY; // [1]

    // Workspace layout (256B aligned slices)
    char* ws = (char*)d_ws;
    size_t off = 0;
    auto alloc = [&](size_t bytes) {
        char* p = ws + off;
        off += (bytes + 255) & ~(size_t)255;
        return p;
    };
    float*    ent_rep  = (float*)   alloc((size_t)B_SZ * D_DIM * sizeof(float));
    float*    word_rep = (float*)   alloc((size_t)B_SZ * D_DIM * sizeof(float));
    _Float16* A_h      = (_Float16*)alloc((size_t)B_SZ * D_DIM * sizeof(_Float16));
    _Float16* E_h      = (_Float16*)alloc((size_t)N_ENTITY * D_DIM * sizeof(_Float16));
    float*    partial  = (float*)   alloc((size_t)B_SZ * sizeof(float));
    (void)ws_size;

    // 1) gather + mean pool
    pool_kernel<<<B_SZ, 256, 0, stream>>>(entity_emb, word_emb,
                                          related_entity, related_word,
                                          ent_rep, word_rep);

    // 2) gate + user (fp32), emit f16 A matrix
    gate_user_kernel<<<B_SZ, 256, 0, stream>>>(ent_rep, word_rep,
                                               w1, b1, w2, b2, A_h);

    // 3) entity_emb -> f16
    {
        const int n4 = (N_ENTITY * D_DIM) / 4;   // 3.2M
        cvt_f16_kernel<<<(n4 + 255) / 256, 256, 0, stream>>>(entity_emb, E_h, n4);
    }

    // 4) WMMA GEMM with TDM-staged B: 16 M-blocks * 625 N-groups = 10000 blocks
    {
        const int blocks = M_BLOCKS * N_GROUPS;  // 10000
        wmma_gemm_kernel<<<blocks, 256, LDS_BYTES, stream>>>(A_h, E_h,
                                                             rec_bias, out_scores);
    }

    // 5) per-row softmax loss partials
    loss_row_kernel<<<B_SZ, 256, 0, stream>>>(out_scores, labels, partial);

    // 6) final mean
    loss_reduce_kernel<<<1, 256, 0, stream>>>(partial, out_loss);

    (void)in_sizes; (void)n_in; (void)out_size;
}
